// MPSOptimizedMultiHeadAttention_6133213298858
// MI455X (gfx1250) — matrix operand order repaired and checked by reading, not compiled, not run on hardware
//
#include <hip/hip_runtime.h>

typedef __attribute__((ext_vector_type(16))) _Float16 v16h;
typedef __attribute__((ext_vector_type(8)))  _Float16 v8h;
typedef __attribute__((ext_vector_type(4)))  _Float16 h4;
typedef __attribute__((ext_vector_type(2)))  _Float16 h2;
typedef __attribute__((ext_vector_type(8)))  float    v8f;
typedef __attribute__((ext_vector_type(2)))  float    f32x2;
typedef __attribute__((ext_vector_type(4)))  float    f32x4;

#define T_SEQ   4096
#define DHEAD   128
#define QTILE   128
#define KVTILE  128
#define NEG_INF (-3.0e38f)

union V16U { v16h h16; v8h h8[2]; };

// One 128-key KV tile step of flash attention (softmax in log2 domain).
template<bool MASKED>
__device__ __forceinline__ void kv_step(
    int j, int jmax, int qs, int wv, int lane, int hi, int l15, int tid,
    const float* __restrict__ Kg, const float* __restrict__ Vg,
    _Float16 (*KT)[KVTILE], _Float16 (*Vs)[DHEAD], _Float16 (*Psw)[KVTILE],
    const v16h (&qf)[4], v8f (&acc)[8], float (&mrow)[8], float (&lrow)[8])
{
    // ---- cooperative K/V tile load: 128 keys x 128 d, f32 -> f16 -----------
    // Each thread owns an adjacent ROW PAIR so transposed K stores pack
    // (key,key+1) into one dword: v_cvt_pk_f16_f32 + ds_store_b32.
    const int rp  = tid >> 2;                 // row pair 0..63 -> rows 2rp,2rp+1
    const int sub = tid & 3;                  // 0..3
    const float* kpa = Kg + (size_t)(j * KVTILE + 2 * rp) * DHEAD;
    const float* kpb = kpa + DHEAD;
    const float* vpa = Vg + (size_t)(j * KVTILE + 2 * rp) * DHEAD;
    const float* vpb = vpa + DHEAD;
    #pragma unroll
    for (int i = 0; i < 8; ++i) {
        const int d = (sub + 4 * i) * 4;      // 0..124, step 4
        f32x4 ka = *(const f32x4*)(kpa + d);
        f32x4 kb = *(const f32x4*)(kpb + d);
        f32x4 va = *(const f32x4*)(vpa + d);
        f32x4 vb = *(const f32x4*)(vpb + d);
        #pragma unroll
        for (int c = 0; c < 4; ++c)
            *(h2*)&KT[d + c][2 * rp] = (h2){(_Float16)ka[c], (_Float16)kb[c]};
        *(h4*)&Vs[2 * rp + 0][d] =
            (h4){(_Float16)va.x, (_Float16)va.y, (_Float16)va.z, (_Float16)va.w};
        *(h4*)&Vs[2 * rp + 1][d] =
            (h4){(_Float16)vb.x, (_Float16)vb.y, (_Float16)vb.z, (_Float16)vb.w};
    }
    __syncthreads();

    // ---- prefetch next KV tile into near cache while we compute ------------
    if (j + 1 < jmax) {
        const float* kn = kpa + KVTILE * DHEAD;
        const float* vn = vpa + KVTILE * DHEAD;
        __builtin_prefetch(kn + sub * 32, 0, 3);            // row 2rp lines
        __builtin_prefetch(kn + DHEAD + sub * 32, 0, 3);    // row 2rp+1 lines
        __builtin_prefetch(vn + sub * 32, 0, 3);
        __builtin_prefetch(vn + DHEAD + sub * 32, 0, 3);
    }

    // ---- S = Q * K^T : 8 key sub-tiles x 4 d-chunks of WMMA ----------------
    v8f S[8];
    #pragma unroll
    for (int nt = 0; nt < 8; ++nt) {
        v8f s;
        #pragma unroll
        for (int r = 0; r < 8; ++r) s[r] = 0.0f;
        #pragma unroll
        for (int c = 0; c < 4; ++c) {
            // B frag: n = l15 (= key in sub-tile), k (= d within chunk) in table order over elements
            V16U b;
            #pragma unroll
            for (int v = 0; v < 8; ++v) {
                const int k = ((v < 4) ? (2 * v) : (16 + 2 * (v - 4))) + hi * 8;
                b.h16[2 * v]     = KT[c * 32 + k][nt * 16 + l15];
                b.h16[2 * v + 1] = KT[c * 32 + k + 1][nt * 16 + l15];
            }
            s = __builtin_amdgcn_wmma_f32_16x16x32_f16(
                    false, qf[c], false, b.h16, (short)0, s, false, false);
        }
        if (MASKED) {
            // causal mask: C layout row m = r + hi*8, col n = l15
            const int kg = j * KVTILE + nt * 16 + l15;
            const int q0 = qs + wv * 16 + hi * 8;
            #pragma unroll
            for (int r = 0; r < 8; ++r)
                s[r] = (kg <= q0 + r) ? s[r] : NEG_INF;
        }
        S[nt] = s;
    }

    // ---- online softmax (log2 domain) over this 128-key tile ---------------
    v8f av;   // per-row rescale factors, vectorized for packed f32 muls
    #pragma unroll
    for (int r = 0; r < 8; ++r) {
        float mx = S[0][r];
        #pragma unroll
        for (int nt = 1; nt < 8; ++nt) mx = fmaxf(mx, S[nt][r]);
        mx = fmaxf(mx, __shfl_xor(mx, 1, 32));
        mx = fmaxf(mx, __shfl_xor(mx, 2, 32));
        mx = fmaxf(mx, __shfl_xor(mx, 4, 32));
        mx = fmaxf(mx, __shfl_xor(mx, 8, 32));
        const float mnew  = fmaxf(mrow[r], mx);
        const float alpha = exp2f(mrow[r] - mnew);
        mrow[r] = mnew;
        lrow[r] *= alpha;
        av[r] = alpha;
    }
    #pragma unroll
    for (int nd = 0; nd < 8; ++nd) acc[nd] *= av;

    // ---- P = exp2(S - m); stage to LDS for C->A layout conversion ----------
    #pragma unroll
    for (int nt = 0; nt < 8; ++nt) {
        #pragma unroll
        for (int r = 0; r < 8; ++r) {
            const float p = exp2f(S[nt][r] - mrow[r]);
            lrow[r] += p;                       // per-lane partial row-sum
            Psw[r + hi * 8][nt * 16 + l15] = (_Float16)p;
        }
    }

    // ---- O += P * V : 4 key-chunks of 32 x 8 d-tiles -----------------------
    #pragma unroll
    for (int kt = 0; kt < 4; ++kt) {
        v16h a;
        const int kbase = hi * 8;
        #pragma unroll
        for (int v = 0; v < 8; ++v) {
            const int k = ((v < 4) ? (2 * v) : (16 + 2 * (v - 4))) + kbase;
            a[2 * v]     = Psw[l15][kt * 32 + k];
            a[2 * v + 1] = Psw[l15][kt * 32 + k + 1];
        }
        #pragma unroll
        for (int nd = 0; nd < 8; ++nd) {
            // B frag: n = l15 (= d column), k (= key within chunk) in table order over elements
            V16U b;
            #pragma unroll
            for (int v = 0; v < 8; ++v) {
                const int k = ((v < 4) ? (2 * v) : (16 + 2 * (v - 4))) + kbase;
                b.h16[2 * v]     = Vs[kt * 32 + k][nd * 16 + l15];
                b.h16[2 * v + 1] = Vs[kt * 32 + k + 1][nd * 16 + l15];
            }
            acc[nd] = __builtin_amdgcn_wmma_f32_16x16x32_f16(
                    false, a, false, b.h16, (short)0, acc[nd], false, false);
        }
    }
    __syncthreads();
}

__global__ __launch_bounds__(256) void fa_causal_wmma_kernel(
    const float* __restrict__ Q, const float* __restrict__ K,
    const float* __restrict__ V, float* __restrict__ O)
{
    // K tile transposed [d][key]  -> B-frag for QK^T (k-dim = d)      32 KB
    __shared__ _Float16 KT[DHEAD][KVTILE];
    // V tile row-major  [key][d]  -> B-frag for P*V  (k-dim = key)    32 KB
    __shared__ _Float16 Vs[KVTILE][DHEAD];
    // per-wave P staging: C-layout -> A-layout conversion             32 KB
    __shared__ _Float16 Ps[8][16][KVTILE];

    const int tid  = threadIdx.x;
    const int lane = tid & 31;
    const int wv   = tid >> 5;        // wave id, 0..7
    const int hi   = lane >> 4;       // 0 or 1 (half of wave)
    const int l15  = lane & 15;

    const int qt = blockIdx.x;
    const int bh = blockIdx.y;
    const int qs = qt * QTILE;

    const size_t baseBH = (size_t)bh * T_SEQ * DHEAD;
    // softmax runs in log2 domain: fold log2(e) into the 1/sqrt(D) scale
    const float scale = 0.08838834764831845f * 1.4426950408889634f;

    // ---------------- Q fragments (A layout, 16x32 f16, 4 d-chunks) --------
    // A layout (16-bit, 16x32): lane m = lane&15; element h=2v+s maps to
    // k = (v<4 ? 2v : 16+2(v-4)) + s + (lane<16 ? 0 : 8)
    v16h qf[4];
    {
        const int qrow  = qs + wv * 16 + l15;
        const float* qp = Q + baseBH + (size_t)qrow * DHEAD;
        const int kbase = hi * 8;
        #pragma unroll
        for (int c = 0; c < 4; ++c) {
            #pragma unroll
            for (int v = 0; v < 8; ++v) {
                const int k = ((v < 4) ? (2 * v) : (16 + 2 * (v - 4))) + kbase;
                const int d = c * 32 + k;
                f32x2 x = *(const f32x2*)(qp + d);
                qf[c][2 * v]     = (_Float16)(x.x * scale);
                qf[c][2 * v + 1] = (_Float16)(x.y * scale);
            }
        }
    }

    // ---------------- accumulators & softmax state --------------------------
    v8f acc[8];
    #pragma unroll
    for (int i = 0; i < 8; ++i)
        #pragma unroll
        for (int r = 0; r < 8; ++r) acc[i][r] = 0.0f;

    float mrow[8], lrow[8];
    #pragma unroll
    for (int r = 0; r < 8; ++r) { mrow[r] = NEG_INF; lrow[r] = 0.0f; }

    const float* Kg = K + baseBH;
    const float* Vg = V + baseBH;
    const int jmax  = qt + 1;          // 128-key tiles covering [0, qs+128)
    const int jdiag = qt;              // the single diagonal tile

    // main loop: tiles strictly below the diagonal, no mask code at all
    for (int j = 0; j < jdiag; ++j)
        kv_step<false>(j, jmax, qs, wv, lane, hi, l15, tid, Kg, Vg,
                       KT, Vs, Ps[wv], qf, acc, mrow, lrow);
    // tail: the diagonal tile with causal masking
    kv_step<true>(jdiag, jmax, qs, wv, lane, hi, l15, tid, Kg, Vg,
                  KT, Vs, Ps[wv], qf, acc, mrow, lrow);

    // ---------------- epilogue: reduce l across lanes, normalize, store ----
    #pragma unroll
    for (int r = 0; r < 8; ++r) {
        float l = lrow[r];
        l += __shfl_xor(l, 1, 32);
        l += __shfl_xor(l, 2, 32);
        l += __shfl_xor(l, 4, 32);
        l += __shfl_xor(l, 8, 32);
        lrow[r] = 1.0f / l;
    }
    float* op = O + baseBH;
    #pragma unroll
    for (int nd = 0; nd < 8; ++nd) {
        #pragma unroll
        for (int r = 0; r < 8; ++r) {
            const int qrow = qs + wv * 16 + r + hi * 8;
            const int d    = nd * 16 + l15;
            op[(size_t)qrow * DHEAD + d] = acc[nd][r] * lrow[r];
        }
    }
}

extern "C" void kernel_launch(void* const* d_in, const int* in_sizes, int n_in,
                              void* d_out, int out_size, void* d_ws, size_t ws_size,
                              hipStream_t stream) {
    const float* Q = (const float*)d_in[0];
    const float* K = (const float*)d_in[1];
    const float* V = (const float*)d_in[2];
    float* Out = (float*)d_out;

    const int BH = in_sizes[0] / (T_SEQ * DHEAD);   // B*H = 32
    dim3 grid(T_SEQ / QTILE, BH);
    fa_causal_wmma_kernel<<<grid, 256, 0, stream>>>(Q, K, V, Out);
}
